// MLayerBiGRU_65515431133357
// MI455X (gfx1250) — compile-verified, matrix operand order checked by reading, not run on hardware
//
#include <hip/hip_runtime.h>
#include <hip/hip_bf16.h>

// MLayerBiGRU for MI455X (gfx1250, wave32, WMMA).
// bf16 WMMA (fp32 accum) everywhere; recurrence partitioned over batch
// (16 rows/WG, zero cross-WG sync); per-step 16x768x256 WMMA across 16 waves;
// xg gate inputs register-prefetched ahead of the WMMA phase and
// global_prefetch'ed one step ahead.

#define B_SZ  256
#define T_SZ  256
#define NIN_SZ 128
#define H_SZ  256
#define G3    768   // 3*H
#define D2    512   // 2*H
#define HGS   772   // hg LDS row stride (768 + 4 pad -> row stride % 64 banks != 0)

typedef __attribute__((ext_vector_type(16))) __bf16 bf16x16;
typedef __attribute__((ext_vector_type(8)))  __bf16 bf16x8;
typedef __attribute__((ext_vector_type(8)))  float  f32x8;

// Load one 16x32 (rows x K) bf16 WMMA operand fragment from a row-major
// matrix with leading dim ld. Works for A (rows = M) and for B loaded as
// W^T from row-major W[N,K] (rows = N). Layout per CDNA5 ISA 7.12.2:
//   lane r = row (lane&15); elems 0..7 -> K = (lane>=16?8:0)+i,
//   elems 8..15 -> K = 16+(lane>=16?8:0)+(i-8)  => two contiguous 16B loads.
__device__ __forceinline__ bf16x16 load_frag16(const __bf16* __restrict__ p, int ld) {
  const int lane = threadIdx.x & 31;
  const int r  = lane & 15;
  const int kb = (lane >> 4) << 3;
  const __bf16* q = p + (size_t)r * ld + kb;
  union { bf16x16 v; bf16x8 h[2]; } u;
  u.h[0] = *(const bf16x8*)(q);
  u.h[1] = *(const bf16x8*)(q + 16);
  return u.v;
}

// ---------------------------------------------------------------------------
// Batched fp32 -> bf16 convert: segment per blockIdx.y (one launch total).
// ---------------------------------------------------------------------------
#define NSEG 9
struct CvtBatch {
  const float* src[NSEG];
  __bf16*      dst[NSEG];
  unsigned long long n[NSEG];
};

__global__ void __launch_bounds__(256) cvt_batch(CvtBatch cb) {
  const int seg = blockIdx.y;
  const float* __restrict__ s = cb.src[seg];
  __bf16* __restrict__ d = cb.dst[seg];
  const size_t n = cb.n[seg];
  size_t i  = (size_t)blockIdx.x * blockDim.x + threadIdx.x;
  const size_t st = (size_t)gridDim.x * blockDim.x;
  for (; i < n; i += st) d[i] = (__bf16)s[i];
}

// ---------------------------------------------------------------------------
// C[M,N] = A[M,K] @ W[N,K]^T + bias[N]  (bf16 in, fp32 out), both directions
// fused via blockIdx.z. block = 128 (4 waves); block tile 64x64; wave 16x64.
// ---------------------------------------------------------------------------
__global__ void __launch_bounds__(128) wmma_gemm_bias2(
    const __bf16* __restrict__ A,
    const __bf16* __restrict__ Wf, const __bf16* __restrict__ Wr,
    const float* __restrict__ biasf, const float* __restrict__ biasr,
    float* __restrict__ Cf, float* __restrict__ Cr,
    int M, int N, int K)
{
  const __bf16* __restrict__ W   = blockIdx.z ? Wr : Wf;
  const float*  __restrict__ bias = blockIdx.z ? biasr : biasf;
  float*        __restrict__ C   = blockIdx.z ? Cr : Cf;

  const int wave = threadIdx.x >> 5;
  const int row0 = blockIdx.x * 64 + wave * 16;
  const int col0 = blockIdx.y * 64;

  f32x8 acc0 = {}, acc1 = {}, acc2 = {}, acc3 = {};
  for (int kk = 0; kk < K; kk += 32) {
    bf16x16 a  = load_frag16(A + (size_t)row0 * K + kk, K);
    bf16x16 b0 = load_frag16(W + (size_t)(col0 +  0) * K + kk, K);
    bf16x16 b1 = load_frag16(W + (size_t)(col0 + 16) * K + kk, K);
    bf16x16 b2 = load_frag16(W + (size_t)(col0 + 32) * K + kk, K);
    bf16x16 b3 = load_frag16(W + (size_t)(col0 + 48) * K + kk, K);
    acc0 = __builtin_amdgcn_wmma_f32_16x16x32_bf16(false, a, false, b0, (short)0, acc0, false, false);
    acc1 = __builtin_amdgcn_wmma_f32_16x16x32_bf16(false, a, false, b1, (short)0, acc1, false, false);
    acc2 = __builtin_amdgcn_wmma_f32_16x16x32_bf16(false, a, false, b2, (short)0, acc2, false, false);
    acc3 = __builtin_amdgcn_wmma_f32_16x16x32_bf16(false, a, false, b3, (short)0, acc3, false, false);
  }

  const int lane = threadIdx.x & 31;
  const int nl = lane & 15;
  const int mo = (lane >> 4) << 3;   // C/D: M = vgpr_idx + 8*(lane>=16)
  f32x8 accs[4] = {acc0, acc1, acc2, acc3};
#pragma unroll
  for (int j = 0; j < 4; ++j) {
    const int n = col0 + j * 16 + nl;
    const float bb = bias[n];
#pragma unroll
    for (int i = 0; i < 8; ++i)
      C[(size_t)(row0 + mo + i) * N + n] = accs[j][i] + bb;
  }
}

// ---------------------------------------------------------------------------
// GRU recurrent scan. grid = (B/16, 2 directions), block = 512 (16 waves,
// 4 per SIMD to hide WMMA/L2 dependency latency). Each block owns 16 batch
// rows for all T. Per step: xg register-prefetch -> hg = h@whh^T (24 WMMA
// per wave) -> fused gates; next step's xg prefetched into cache.
// ---------------------------------------------------------------------------
__global__ void __launch_bounds__(512) gru_scan(
    const float* __restrict__ xg_f, const float* __restrict__ xg_r,
    const __bf16* __restrict__ whh_f, const __bf16* __restrict__ whh_r,
    const float* __restrict__ bhh_f, const float* __restrict__ bhh_r,
    float* __restrict__ out_f32, __bf16* __restrict__ out_bf,
    float* __restrict__ hfin_f, float* __restrict__ hfin_r)
{
  const int  dir = blockIdx.y;
  const bool rev = (dir == 1);
  const float*  __restrict__ xg  = rev ? xg_r  : xg_f;
  const __bf16* __restrict__ whh = rev ? whh_r : whh_f;
  const float*  __restrict__ bhh = rev ? bhh_r : bhh_f;
  float* hfin = rev ? hfin_r : hfin_f;
  const int out_off = dir * H_SZ;
  const int b0  = blockIdx.x * 16;
  const int tid = threadIdx.x;

  __shared__ __align__(16) __bf16 h_bf[16 * H_SZ];   // 8 KB
  __shared__ __align__(16) float  hg  [16 * HGS];    // ~48.3 KB (padded stride)
  __shared__ float bhh_s[G3];                        // 3 KB

  const int c  = tid & 255;          // hidden column owned by this thread
  const int rh = (tid >> 8) << 3;    // row-half base: 0 or 8

  float hreg[8];                     // fp32 state for rows rh..rh+7, col c
#pragma unroll
  for (int i = 0; i < 8; ++i) hreg[i] = 0.f;
  for (int i = tid; i < 16 * H_SZ; i += 512) h_bf[i] = (__bf16)0.f;
  for (int i = tid; i < G3; i += 512) bhh_s[i] = bhh[i];
  __syncthreads();

  const int wave = tid >> 5;
  const int lane = tid & 31;
  const int nl = lane & 15;
  const int mo = (lane >> 4) << 3;

  for (int t = 0; t < T_SZ; ++t) {
    const int tt = rev ? (T_SZ - 1 - t) : t;

    // ---- xg for this step into registers; overlaps the whole WMMA phase ----
    float pxr[8], pxz[8], pxn[8];
#pragma unroll
    for (int i = 0; i < 8; ++i) {
      const size_t xrow = ((size_t)(b0 + rh + i) * T_SZ + tt) * G3;
      pxr[i] = xg[xrow + c];
      pxz[i] = xg[xrow + H_SZ + c];
      pxn[i] = xg[xrow + 2 * H_SZ + c];
    }

    // ---- hg = h @ whh^T ----
    bf16x16 afr[8];
#pragma unroll
    for (int k8 = 0; k8 < 8; ++k8)
      afr[k8] = load_frag16(h_bf + k8 * 32, H_SZ);

#pragma unroll
    for (int jt = 0; jt < 3; ++jt) {
      const int n0 = (wave * 3 + jt) * 16;
      f32x8 acc = {};
#pragma unroll
      for (int k8 = 0; k8 < 8; ++k8) {
        bf16x16 bfr = load_frag16(whh + (size_t)n0 * H_SZ + k8 * 32, H_SZ);
        acc = __builtin_amdgcn_wmma_f32_16x16x32_bf16(false, afr[k8], false, bfr, (short)0, acc, false, false);
      }
      const int n = n0 + nl;
#pragma unroll
      for (int i = 0; i < 8; ++i) hg[(mo + i) * HGS + n] = acc[i];
    }

    // ---- prefetch next step's xg rows into cache (global_prefetch_b8) ----
    if (t + 1 < T_SZ) {
      const int tt2 = rev ? (tt - 1) : (tt + 1);
#pragma unroll
      for (int i = 0; i < 8; ++i) {
        const size_t xrow2 = ((size_t)(b0 + rh + i) * T_SZ + tt2) * G3;
        // 256 threads/row * 12B stride covers the full 3072B row (every line)
        __builtin_prefetch((const char*)(xg + xrow2) + (size_t)c * 12, 0, 3);
      }
    }
    __syncthreads();

    // ---- fused gates: thread owns col c, rows rh..rh+7 ----
#pragma unroll
    for (int i = 0; i < 8; ++i) {
      const int row = rh + i;
      const float hr = hg[row * HGS + c]            + bhh_s[c];
      const float hz = hg[row * HGS + H_SZ + c]     + bhh_s[H_SZ + c];
      const float hn = hg[row * HGS + 2 * H_SZ + c] + bhh_s[2 * H_SZ + c];
      const float r = 1.f / (1.f + __expf(-(pxr[i] + hr)));
      const float z = 1.f / (1.f + __expf(-(pxz[i] + hz)));
      const float n = tanhf(pxn[i] + r * hn);
      const float hnew = (1.f - z) * n + z * hreg[i];
      hreg[i] = hnew;
      h_bf[row * H_SZ + c] = (__bf16)hnew;
      const size_t orow = ((size_t)(b0 + row) * T_SZ + tt) * D2 + out_off;
      if (out_bf) out_bf[orow + c] = (__bf16)hnew;
      else        out_f32[orow + c] = hnew;
    }
    __syncthreads();
  }

  if (hfin) {
#pragma unroll
    for (int i = 0; i < 8; ++i)
      hfin[(size_t)(b0 + rh + i) * H_SZ + c] = hreg[i];
  }
}

// ---------------------------------------------------------------------------
// Attention + output: one block per batch row.
// ---------------------------------------------------------------------------
__global__ void __launch_bounds__(256) attn_out(
    const float* __restrict__ rec2, const float* __restrict__ h2f,
    const float* __restrict__ h2r, float* __restrict__ out)
{
  const int b = blockIdx.x;
  const int tid = threadIdx.x;
  __shared__ __align__(16) float merged[D2];
  __shared__ float sc[T_SZ];
  __shared__ float red[256];

  merged[tid]        = h2f[(size_t)b * H_SZ + tid];
  merged[H_SZ + tid] = h2r[(size_t)b * H_SZ + tid];
  __syncthreads();

  // score for timestep t = tid
  const float4* row = (const float4*)(rec2 + ((size_t)b * T_SZ + tid) * D2);
  const float4* mg  = (const float4*)merged;
  float s = 0.f;
#pragma unroll 4
  for (int q = 0; q < D2 / 4; ++q) {
    const float4 rv = row[q], mv = mg[q];
    s += rv.x * mv.x + rv.y * mv.y + rv.z * mv.z + rv.w * mv.w;
  }

  red[tid] = s; __syncthreads();
  for (int off = 128; off > 0; off >>= 1) {
    if (tid < off) red[tid] = fmaxf(red[tid], red[tid + off]);
    __syncthreads();
  }
  const float mx = red[0]; __syncthreads();
  const float e = __expf(s - mx);
  sc[tid] = e;
  red[tid] = e; __syncthreads();
  for (int off = 128; off > 0; off >>= 1) {
    if (tid < off) red[tid] += red[tid + off];
    __syncthreads();
  }
  const float inv = 1.f / red[0];
  __syncthreads();

  float a0 = 0.f, a1 = 0.f;
  for (int t = 0; t < T_SZ; ++t) {
    const float w = sc[t];
    const float* r = rec2 + ((size_t)b * T_SZ + t) * D2;
    a0 += w * r[tid];
    a1 += w * r[H_SZ + tid];
  }
  float* ob = out + (size_t)b * (4 * H_SZ);
  ob[tid]            = a0 * inv;
  ob[H_SZ + tid]     = a1 * inv;
  const float* last = rec2 + ((size_t)b * T_SZ + (T_SZ - 1)) * D2;
  ob[D2 + tid]        = last[tid];
  ob[D2 + H_SZ + tid] = last[H_SZ + tid];
}

// ---------------------------------------------------------------------------
extern "C" void kernel_launch(void* const* d_in, const int* in_sizes, int n_in,
                              void* d_out, int out_size, void* d_ws, size_t ws_size,
                              hipStream_t stream) {
  (void)in_sizes; (void)n_in; (void)out_size; (void)ws_size;
  const float* input = (const float*)d_in[0];
  const float* wih1f = (const float*)d_in[1];
  const float* whh1f = (const float*)d_in[2];
  const float* bih1f = (const float*)d_in[3];
  const float* bhh1f = (const float*)d_in[4];
  const float* wih1r = (const float*)d_in[5];
  const float* whh1r = (const float*)d_in[6];
  const float* bih1r = (const float*)d_in[7];
  const float* bhh1r = (const float*)d_in[8];
  const float* wih2f = (const float*)d_in[9];
  const float* whh2f = (const float*)d_in[10];
  const float* bih2f = (const float*)d_in[11];
  const float* bhh2f = (const float*)d_in[12];
  const float* wih2r = (const float*)d_in[13];
  const float* whh2r = (const float*)d_in[14];
  const float* bih2r = (const float*)d_in[15];
  const float* bhh2r = (const float*)d_in[16];

  char* ws = (char*)d_ws;
  size_t off = 0;
  auto alloc = [&](size_t bytes) -> void* {
    void* p = ws + off;
    off = (off + bytes + 255) & ~(size_t)255;
    return p;
  };

  const size_t MT = (size_t)B_SZ * T_SZ;                 // 65536 rows
  __bf16* xbf      = (__bf16*)alloc(MT * NIN_SZ * 2);
  __bf16* wih1f_b  = (__bf16*)alloc((size_t)G3 * NIN_SZ * 2);
  __bf16* wih1r_b  = (__bf16*)alloc((size_t)G3 * NIN_SZ * 2);
  __bf16* whh1f_b  = (__bf16*)alloc((size_t)G3 * H_SZ * 2);
  __bf16* whh1r_b  = (__bf16*)alloc((size_t)G3 * H_SZ * 2);
  __bf16* wih2f_b  = (__bf16*)alloc((size_t)G3 * D2 * 2);
  __bf16* wih2r_b  = (__bf16*)alloc((size_t)G3 * D2 * 2);
  __bf16* whh2f_b  = (__bf16*)alloc((size_t)G3 * H_SZ * 2);
  __bf16* whh2r_b  = (__bf16*)alloc((size_t)G3 * H_SZ * 2);
  __bf16* rec1_bf  = (__bf16*)alloc(MT * D2 * 2);
  float*  rec2     = (float*)alloc(MT * D2 * 4);
  float*  h2f      = (float*)alloc((size_t)B_SZ * H_SZ * 4);
  float*  h2r      = (float*)alloc((size_t)B_SZ * H_SZ * 4);
  float*  xgf      = (float*)alloc(MT * G3 * 4);          // reused layer1+layer2
  float*  xgr      = (float*)alloc(MT * G3 * 4);

  // --- single batched fp32 -> bf16 conversion launch ---
  CvtBatch cb;
  cb.src[0] = input; cb.dst[0] = xbf;     cb.n[0] = MT * NIN_SZ;
  cb.src[1] = wih1f; cb.dst[1] = wih1f_b; cb.n[1] = (size_t)G3 * NIN_SZ;
  cb.src[2] = wih1r; cb.dst[2] = wih1r_b; cb.n[2] = (size_t)G3 * NIN_SZ;
  cb.src[3] = whh1f; cb.dst[3] = whh1f_b; cb.n[3] = (size_t)G3 * H_SZ;
  cb.src[4] = whh1r; cb.dst[4] = whh1r_b; cb.n[4] = (size_t)G3 * H_SZ;
  cb.src[5] = wih2f; cb.dst[5] = wih2f_b; cb.n[5] = (size_t)G3 * D2;
  cb.src[6] = wih2r; cb.dst[6] = wih2r_b; cb.n[6] = (size_t)G3 * D2;
  cb.src[7] = whh2f; cb.dst[7] = whh2f_b; cb.n[7] = (size_t)G3 * H_SZ;
  cb.src[8] = whh2r; cb.dst[8] = whh2r_b; cb.n[8] = (size_t)G3 * H_SZ;
  cvt_batch<<<dim3(512, NSEG), 256, 0, stream>>>(cb);

  const dim3 ggrid((unsigned)(MT / 64), G3 / 64, 2);      // (1024, 12, 2)

  // --- layer 1: input projections (f+r fused) + scan ---
  wmma_gemm_bias2<<<ggrid, 128, 0, stream>>>(xbf, wih1f_b, wih1r_b, bih1f, bih1r,
                                             xgf, xgr, (int)MT, G3, NIN_SZ);
  gru_scan<<<dim3(B_SZ / 16, 2), 512, 0, stream>>>(
      xgf, xgr, whh1f_b, whh1r_b, bhh1f, bhh1r,
      nullptr, rec1_bf, nullptr, nullptr);

  // --- layer 2: input projections (f+r fused) + scan ---
  wmma_gemm_bias2<<<ggrid, 128, 0, stream>>>(rec1_bf, wih2f_b, wih2r_b, bih2f, bih2r,
                                             xgf, xgr, (int)MT, G3, D2);
  gru_scan<<<dim3(B_SZ / 16, 2), 512, 0, stream>>>(
      xgf, xgr, whh2f_b, whh2r_b, bhh2f, bhh2r,
      rec2, nullptr, h2f, h2r);

  // --- attention + output ---
  attn_out<<<B_SZ, 256, 0, stream>>>(rec2, h2f, h2r, (float*)d_out);
}
